// RosslerApproximator_9457517986371
// MI455X (gfx1250) — compile-verified, matrix operand order checked by reading, not run on hardware
//
#include <hip/hip_runtime.h>
#include <hip/hip_bf16.h>
#include <math.h>

#define BN 131072
#define BE 1048576
#define NGRAPH 16

#define EDGE_TILES (BE / 32)   // 32768 tiles of 32 edges
#define NODE_TILES (BN / 32)   // 4096 tiles of 32 nodes
#define EDGE_GRID 1024         // 4096 waves -> 8 tiles/wave
#define NODE_GRID 512          // 2048 waves -> 2 tiles/wave

typedef __attribute__((ext_vector_type(16))) _Float16 v16h;
typedef __attribute__((ext_vector_type(8)))  _Float16 v8h;
typedef __attribute__((ext_vector_type(8)))  float    v8f;

__device__ __forceinline__ float gelu_f(float x) {
  // jax.nn.gelu default (approximate=True, tanh form)
  return 0.5f * x * (1.0f + tanhf(0.7978845608028654f * (x + 0.044715f * x * x * x)));
}

__device__ __forceinline__ v8f zero8f() {
  v8f z;
  #pragma unroll
  for (int i = 0; i < 8; ++i) z[i] = 0.0f;
  return z;
}

// Load a 16x32 f16 A/B fragment (ISA 7.12.2 layout).
// rowk0 = &M[row][k0] where row = lane&15 selects the matrix row (A: m, B(NxK): n).
// Lanes 0-15 hold k = {0..7, 16..23}, lanes 16-31 hold k = {8..15, 24..31}.
__device__ __forceinline__ v16h load_frag(const _Float16* rowk0, int lane) {
  const _Float16* p = rowk0 + ((lane >> 4) << 3);
  v8h lo = *(const v8h*)p;
  v8h hi = *(const v8h*)(p + 16);
  v16h r;
  #pragma unroll
  for (int i = 0; i < 8; ++i) { r[i] = lo[i]; r[i + 8] = hi[i]; }
  return r;
}

// ---------------- encoders ----------------

__global__ void encode_nodes_kernel(const float* __restrict__ x,
                                    const float* __restrict__ Ws,
                                    const float* __restrict__ bs,
                                    _Float16* __restrict__ nemb) {
  int i = blockIdx.x * 256 + threadIdx.x;         // over BN*48
  if (i >= BN * 48) return;
  int node = i / 48;
  int rem  = i - node * 48;
  int f = rem >> 4;                               // feature 0..2
  int d = rem & 15;                               // emb dim 0..15
  float v = x[node * 3 + f] * Ws[f * 16 + d] + bs[f * 16 + d];
  nemb[i] = (_Float16)gelu_f(v);
}

__global__ void encode_edges_kernel(const float* __restrict__ ea,
                                    const float* __restrict__ We,
                                    const float* __restrict__ be,
                                    _Float16* __restrict__ eemb) {
  long i = (long)blockIdx.x * 256 + threadIdx.x;  // over BE*16
  if (i >= (long)BE * 16) return;
  long e = i >> 4;
  int d = (int)(i & 15);
  float v = ea[e] * We[d] + be[d];
  eemb[i] = (_Float16)gelu_f(v);
}

__global__ void encode_glob_kernel(const float* __restrict__ ga,
                                   const float* __restrict__ Wg,
                                   const float* __restrict__ bg,
                                   _Float16* __restrict__ gemb) {
  int i = threadIdx.x;                            // 768 = 16*48
  if (i >= NGRAPH * 48) return;
  int g = i / 48;
  int rem = i - g * 48;
  int f = rem >> 4;
  int d = rem & 15;
  float v = ga[g * 3 + f] * Wg[f * 16 + d] + bg[f * 16 + d];
  gemb[i] = (_Float16)gelu_f(v);
}

// ---------------- weight prep (transpose to NxK f16, pad, fold Wn2@Wd) ----------------

__global__ void prep_weights_kernel(const float* __restrict__ We1,
                                    const float* __restrict__ We2,
                                    const float* __restrict__ Wn1,
                                    const float* __restrict__ Wn2,
                                    const float* __restrict__ Wd,
                                    const float* __restrict__ bn2,
                                    const float* __restrict__ bd,
                                    _Float16* __restrict__ W1T,   // 128 x 128 (K padded from 112)
                                    _Float16* __restrict__ W2T,   // 16 x 128
                                    _Float16* __restrict__ Wn1T,  // 128 x 64
                                    _Float16* __restrict__ W2dT,  // 16 x 128 (only n<3 nonzero)
                                    float* __restrict__ bn2d) {   // 3
  int i = blockIdx.x * 256 + threadIdx.x;
  if (i < 128 * 128) {
    int nn = i >> 7, k = i & 127;
    W1T[i] = (_Float16)((k < 112) ? We1[k * 128 + nn] : 0.0f);
  }
  int j = i - 128 * 128;
  if (j >= 0 && j < 16 * 128) {
    int nn = j >> 7, k = j & 127;
    W2T[j] = (_Float16)We2[k * 16 + nn];
  }
  int l = i - 128 * 128 - 16 * 128;
  if (l >= 0 && l < 128 * 64) {
    int nn = l >> 6, k = l & 63;
    Wn1T[l] = (_Float16)Wn1[k * 128 + nn];
  }
  int m = i - 128 * 128 - 16 * 128 - 128 * 64;
  if (m >= 0 && m < 16 * 128) {
    int nn = m >> 7, k = m & 127;
    float s = 0.0f;
    if (nn < 3) {
      for (int jj = 0; jj < 48; ++jj) s += Wn2[k * 48 + jj] * Wd[jj * 3 + nn];
    }
    W2dT[m] = (_Float16)s;
  }
  if (i < 3) {
    float s = bd[i];
    for (int jj = 0; jj < 48; ++jj) s += bn2[jj] * Wd[jj * 3 + i];
    bn2d[i] = s;
  }
}

// ---------------- edge MLP: gather -> 112->128 GELU -> 128->16 -> atomic scatter ----------------
// 4 waves/block; each wave owns a 32-edge (M=32) tile per iteration and grid-strides
// over tiles so the LDS-resident weights are loaded once per block, not per tile.

__global__ __launch_bounds__(128) void edge_mlp_kernel(
    const _Float16* __restrict__ nemb, const _Float16* __restrict__ eemb,
    const int* __restrict__ eidx,
    const _Float16* __restrict__ W1T, const _Float16* __restrict__ W2T,
    const float* __restrict__ be1, const float* __restrict__ be2,
    float* __restrict__ agg) {
  __shared__ _Float16 sW1[128][128];   // 32 KB  W1^T [n][k]
  __shared__ _Float16 sW2[16][128];    //  4 KB  W2^T [n][k]
  __shared__ _Float16 sA[4][32][128];  // 32 KB  per-wave A / H staging (M=32)
  __shared__ int      sDst[4][32];     // destination node cache for the scatter

  const int t = threadIdx.x;
  const int lane = t & 31;
  const int w = t >> 5;

  // cooperative weight load (once per block)
  {
    const v8h* g1 = (const v8h*)W1T;
    v8h* s1 = (v8h*)(&sW1[0][0]);
    for (int i = t; i < (128 * 128) / 8; i += 128) s1[i] = g1[i];
    const v8h* g2 = (const v8h*)W2T;
    v8h* s2 = (v8h*)(&sW2[0][0]);
    for (int i = t; i < (16 * 128) / 8; i += 128) s2[i] = g2[i];
  }

  const int n  = lane & 15;          // D column / A row selector
  const int mg = (lane >> 4) << 3;   // D row group base

  // loop-invariant biases -> registers
  float bias1[8];
  #pragma unroll
  for (int nt = 0; nt < 8; ++nt) bias1[nt] = be1[nt * 16 + n];
  const float bias2 = be2[n];

  const int waveId = blockIdx.x * 4 + w;
  const int waveStride = EDGE_GRID * 4;

  __syncthreads();

  for (int tile = waveId; tile < EDGE_TILES; tile += waveStride) {
    const int e = tile * 32 + lane;

    // prefetch next iteration's streams toward L2 (global_prefetch_b8)
    {
      const int eNext = e + waveStride * 32;
      if (eNext < BE) {
        __builtin_prefetch(eidx + eNext, 0, 0);
        __builtin_prefetch(eidx + BE + eNext, 0, 0);
        __builtin_prefetch(eemb + (long)eNext * 16, 0, 0);
      }
    }

    // stage A row for this lane's edge: [ nemb[src] 48 | nemb[dst] 48 | eemb 16 | 0 pad 16 ]
    {
      int src = eidx[e];
      int dst = eidx[BE + e];
      sDst[w][lane] = dst;
      _Float16* a = &sA[w][lane][0];
      const v8h* ps = (const v8h*)(nemb + (long)src * 48);
      const v8h* pd = (const v8h*)(nemb + (long)dst * 48);
      #pragma unroll
      for (int i = 0; i < 6; ++i) ((v8h*)a)[i] = ps[i];
      #pragma unroll
      for (int i = 0; i < 6; ++i) ((v8h*)(a + 48))[i] = pd[i];
      const v8h* pe = (const v8h*)(eemb + (long)e * 16);
      ((v8h*)(a + 96))[0] = pe[0];
      ((v8h*)(a + 96))[1] = pe[1];
      v8h z;
      #pragma unroll
      for (int i = 0; i < 8; ++i) z[i] = (_Float16)0.0f;
      ((v8h*)(a + 112))[0] = z;
      ((v8h*)(a + 112))[1] = z;
    }
    __syncthreads();

    // GEMM1: 32x128 = (32x128) x (128x128); two M tiles share each B fragment
    v8f acc[2][8];
    #pragma unroll
    for (int mt = 0; mt < 2; ++mt)
      #pragma unroll
      for (int nt = 0; nt < 8; ++nt) acc[mt][nt] = zero8f();

    #pragma unroll
    for (int kt = 0; kt < 4; ++kt) {
      v16h a0 = load_frag(&sA[w][n][kt * 32], lane);
      v16h a1 = load_frag(&sA[w][16 + n][kt * 32], lane);
      #pragma unroll
      for (int nt = 0; nt < 8; ++nt) {
        v16h b = load_frag(&sW1[nt * 16 + n][kt * 32], lane);
        acc[0][nt] = __builtin_amdgcn_wmma_f32_16x16x32_f16(false, a0, false, b, (short)0,
                                                            acc[0][nt], false, false);
        acc[1][nt] = __builtin_amdgcn_wmma_f32_16x16x32_f16(false, a1, false, b, (short)0,
                                                            acc[1][nt], false, false);
      }
    }
    __syncthreads();

    // bias + GELU, restage hidden as f16 A-matrix (32x128)
    #pragma unroll
    for (int mt = 0; mt < 2; ++mt)
      #pragma unroll
      for (int nt = 0; nt < 8; ++nt) {
        #pragma unroll
        for (int r = 0; r < 8; ++r) {
          sA[w][mt * 16 + mg + r][nt * 16 + n] =
              (_Float16)gelu_f(acc[mt][nt][r] + bias1[nt]);
        }
      }
    __syncthreads();

    // GEMM2: 32x16 = (32x128) x (128x16)
    v8f acc2[2];
    acc2[0] = zero8f();
    acc2[1] = zero8f();
    #pragma unroll
    for (int kt = 0; kt < 4; ++kt) {
      v16h a0 = load_frag(&sA[w][n][kt * 32], lane);
      v16h a1 = load_frag(&sA[w][16 + n][kt * 32], lane);
      v16h b  = load_frag(&sW2[n][kt * 32], lane);
      acc2[0] = __builtin_amdgcn_wmma_f32_16x16x32_f16(false, a0, false, b, (short)0,
                                                       acc2[0], false, false);
      acc2[1] = __builtin_amdgcn_wmma_f32_16x16x32_f16(false, a1, false, b, (short)0,
                                                       acc2[1], false, false);
    }

    // scatter-add to destination nodes (L2-resident agg, 16 features per edge)
    #pragma unroll
    for (int mt = 0; mt < 2; ++mt) {
      #pragma unroll
      for (int r = 0; r < 8; ++r) {
        int dst = sDst[w][mt * 16 + mg + r];
        atomicAdd(&agg[(long)dst * 16 + n], acc2[mt][r] + bias2);
      }
    }
    __syncthreads();   // protect sA/sDst reuse across iterations
  }
}

// ---------------- node MLP: [agg|glob] 64 -> 128 GELU -> (Wn2@Wd) -> 3 ----------------

__global__ __launch_bounds__(128) void node_mlp_kernel(
    const float* __restrict__ agg, const _Float16* __restrict__ gemb,
    const int* __restrict__ batch,
    const _Float16* __restrict__ Wn1T, const _Float16* __restrict__ W2dT,
    const float* __restrict__ bn1, const float* __restrict__ bn2d,
    float* __restrict__ out) {
  __shared__ _Float16 sW1[128][64];    // 16 KB  Wn1^T [n][k]
  __shared__ _Float16 sW2[16][128];    //  4 KB  (Wn2@Wd)^T padded [n][k]
  __shared__ _Float16 sA[4][32][128];  // 32 KB  per-wave input (cols 0..63) / hidden staging

  const int t = threadIdx.x;
  const int lane = t & 31;
  const int w = t >> 5;

  {
    const v8h* g1 = (const v8h*)Wn1T;
    v8h* s1 = (v8h*)(&sW1[0][0]);
    for (int i = t; i < (128 * 64) / 8; i += 128) s1[i] = g1[i];
    const v8h* g2 = (const v8h*)W2dT;
    v8h* s2 = (v8h*)(&sW2[0][0]);
    for (int i = t; i < (16 * 128) / 8; i += 128) s2[i] = g2[i];
  }

  const int n  = lane & 15;
  const int mg = (lane >> 4) << 3;

  float bias1[8];
  #pragma unroll
  for (int nt = 0; nt < 8; ++nt) bias1[nt] = bn1[nt * 16 + n];
  const float biasOut = (n < 3) ? bn2d[n] : 0.0f;

  const int waveId = blockIdx.x * 4 + w;
  const int waveStride = NODE_GRID * 4;

  __syncthreads();

  for (int tile = waveId; tile < NODE_TILES; tile += waveStride) {
    const int nodeBase = tile * 32;

    // stage A row for this lane's node: [ agg (16, f32->f16) | gemb[batch] (48) ]
    {
      int nd = nodeBase + lane;
      _Float16* a = &sA[w][lane][0];
      const float* pa = agg + (long)nd * 16;
      #pragma unroll
      for (int i = 0; i < 16; ++i) a[i] = (_Float16)pa[i];
      int g = batch[nd];
      const v8h* pg = (const v8h*)(gemb + (long)g * 48);
      #pragma unroll
      for (int i = 0; i < 6; ++i) ((v8h*)(a + 16))[i] = pg[i];
    }
    __syncthreads();

    // GEMM1: 32x128 = (32x64) x (64x128)
    v8f acc[2][8];
    #pragma unroll
    for (int mt = 0; mt < 2; ++mt)
      #pragma unroll
      for (int nt = 0; nt < 8; ++nt) acc[mt][nt] = zero8f();

    #pragma unroll
    for (int kt = 0; kt < 2; ++kt) {
      v16h a0 = load_frag(&sA[w][n][kt * 32], lane);
      v16h a1 = load_frag(&sA[w][16 + n][kt * 32], lane);
      #pragma unroll
      for (int nt = 0; nt < 8; ++nt) {
        v16h b = load_frag(&sW1[nt * 16 + n][kt * 32], lane);
        acc[0][nt] = __builtin_amdgcn_wmma_f32_16x16x32_f16(false, a0, false, b, (short)0,
                                                            acc[0][nt], false, false);
        acc[1][nt] = __builtin_amdgcn_wmma_f32_16x16x32_f16(false, a1, false, b, (short)0,
                                                            acc[1][nt], false, false);
      }
    }
    __syncthreads();

    #pragma unroll
    for (int mt = 0; mt < 2; ++mt)
      #pragma unroll
      for (int nt = 0; nt < 8; ++nt) {
        #pragma unroll
        for (int r = 0; r < 8; ++r) {
          sA[w][mt * 16 + mg + r][nt * 16 + n] =
              (_Float16)gelu_f(acc[mt][nt][r] + bias1[nt]);
        }
      }
    __syncthreads();

    // GEMM2: 32x16 = (32x128) x (128x16); only columns 0..2 are meaningful
    v8f acc2[2];
    acc2[0] = zero8f();
    acc2[1] = zero8f();
    #pragma unroll
    for (int kt = 0; kt < 4; ++kt) {
      v16h a0 = load_frag(&sA[w][n][kt * 32], lane);
      v16h a1 = load_frag(&sA[w][16 + n][kt * 32], lane);
      v16h b  = load_frag(&sW2[n][kt * 32], lane);
      acc2[0] = __builtin_amdgcn_wmma_f32_16x16x32_f16(false, a0, false, b, (short)0,
                                                       acc2[0], false, false);
      acc2[1] = __builtin_amdgcn_wmma_f32_16x16x32_f16(false, a1, false, b, (short)0,
                                                       acc2[1], false, false);
    }

    if (n < 3) {
      #pragma unroll
      for (int mt = 0; mt < 2; ++mt) {
        #pragma unroll
        for (int r = 0; r < 8; ++r) {
          int nd = nodeBase + mt * 16 + mg + r;
          out[(long)nd * 3 + n] = acc2[mt][r] + biasOut;
        }
      }
    }
    __syncthreads();
  }
}

// ---------------- launcher ----------------

extern "C" void kernel_launch(void* const* d_in, const int* in_sizes, int n_in,
                              void* d_out, int out_size, void* d_ws, size_t ws_size,
                              hipStream_t stream) {
  (void)in_sizes; (void)n_in; (void)out_size; (void)ws_size;

  const float* x         = (const float*)d_in[0];
  const int*   eidx      = (const int*)d_in[1];
  const int*   batch     = (const int*)d_in[2];
  /* d_in[3] node_attr: zero-width */
  const float* edge_attr = (const float*)d_in[4];
  const float* glob_attr = (const float*)d_in[5];
  const float* Ws  = (const float*)d_in[6];
  const float* bs  = (const float*)d_in[7];
  const float* We  = (const float*)d_in[8];
  const float* be  = (const float*)d_in[9];
  const float* Wg  = (const float*)d_in[10];
  const float* bg  = (const float*)d_in[11];
  const float* We1 = (const float*)d_in[12];
  const float* be1 = (const float*)d_in[13];
  const float* We2 = (const float*)d_in[14];
  const float* be2 = (const float*)d_in[15];
  const float* Wn1 = (const float*)d_in[16];
  const float* bn1 = (const float*)d_in[17];
  const float* Wn2 = (const float*)d_in[18];
  const float* bn2 = (const float*)d_in[19];
  const float* Wd  = (const float*)d_in[20];
  const float* bd  = (const float*)d_in[21];
  float* out = (float*)d_out;

  char* ws = (char*)d_ws;
  size_t off = 0;
  auto alloc = [&](size_t bytes) -> void* {
    void* p = (void*)(ws + off);
    off += (bytes + 255) & ~(size_t)255;
    return p;
  };

  _Float16* nemb  = (_Float16*)alloc((size_t)BN * 48 * 2);
  _Float16* eemb  = (_Float16*)alloc((size_t)BE * 16 * 2);
  _Float16* gemb  = (_Float16*)alloc((size_t)NGRAPH * 48 * 2);
  float*    agg   = (float*)alloc((size_t)BN * 16 * 4);
  _Float16* W1T   = (_Float16*)alloc(128 * 128 * 2);
  _Float16* W2T   = (_Float16*)alloc(16 * 128 * 2);
  _Float16* Wn1T  = (_Float16*)alloc(128 * 64 * 2);
  _Float16* W2dT  = (_Float16*)alloc(16 * 128 * 2);
  float*    bn2d  = (float*)alloc(16);

  hipMemsetAsync(agg, 0, (size_t)BN * 16 * 4, stream);

  prep_weights_kernel<<<112, 256, 0, stream>>>(We1, We2, Wn1, Wn2, Wd, bn2, bd,
                                               W1T, W2T, Wn1T, W2dT, bn2d);
  encode_nodes_kernel<<<(BN * 48) / 256, 256, 0, stream>>>(x, Ws, bs, nemb);
  encode_edges_kernel<<<(int)(((long)BE * 16) / 256), 256, 0, stream>>>(edge_attr, We, be, eemb);
  encode_glob_kernel<<<1, NGRAPH * 48, 0, stream>>>(glob_attr, Wg, bg, gemb);

  edge_mlp_kernel<<<EDGE_GRID, 128, 0, stream>>>(nemb, eemb, eidx, W1T, W2T, be1, be2, agg);
  node_mlp_kernel<<<NODE_GRID, 128, 0, stream>>>(agg, gemb, batch, Wn1T, W2dT, bn1, bn2d, out);
}
